// RelationalTransformer_21088289424019
// MI455X (gfx1250) — matrix operand order repaired and checked by reading, not compiled, not run on hardware
//
#include <hip/hip_runtime.h>
#include <hip/hip_bf16.h>
#include <math.h>
#include <stdint.h>

typedef __attribute__((ext_vector_type(16))) __bf16 v16bf;
typedef __attribute__((ext_vector_type(8)))  __bf16 v8bf;
typedef __attribute__((ext_vector_type(4)))  __bf16 v4bf;
typedef __attribute__((ext_vector_type(8)))  float  v8f;
typedef __attribute__((ext_vector_type(4)))  unsigned int v4u;
typedef __attribute__((ext_vector_type(8)))  int v8i;
typedef __attribute__((ext_vector_type(4)))  int v4i;

#define NND   256      // nodes
#define DIMN  512      // node dim
#define DIME  256      // edge dim
#define NHEAD 8
#define EPS_RMS 1.1920928955078125e-07f
#define EPS_LN  1e-5f

// Tensor Data Mover availability (ROCm7.2 clang-22 = 5-arg; clang>=23 = 6-arg)
#if defined(__has_builtin)
# if __has_builtin(__builtin_amdgcn_tensor_load_to_lds) && __has_builtin(__builtin_amdgcn_s_wait_tensorcnt)
#  define HAVE_TDM 1
# endif
#endif
#ifndef HAVE_TDM
# define HAVE_TDM 0
#endif

// ---------------------------------------------------------------- embedding
__global__ void embed_kernel(const int* __restrict__ x, const float* __restrict__ tok,
                             const float* __restrict__ pos, float* __restrict__ h) {
  int idx = blockIdx.x * 256 + threadIdx.x;
  int n = idx / DIMN, d = idx % DIMN;
  h[idx] = tok[(long)x[n] * DIMN + d] + pos[(long)n * DIMN + d];
}

// ---------------------------------------------------------------- f32 -> bf16
__global__ void cvt_bf16_kernel(const float* __restrict__ in, __bf16* __restrict__ out, long n) {
  long i = (long)blockIdx.x * 256 + threadIdx.x;
  if (i < n) out[i] = (__bf16)in[i];
}

// ---------------------------------------------------------------- rmsnorm -> bf16 (1 block/row)
__global__ void rmsnorm_kernel(const float* __restrict__ in, const float* __restrict__ w,
                               __bf16* __restrict__ out, int cols, float eps) {
  __shared__ float red[256];
  long row = blockIdx.x;
  int t = threadIdx.x;
  const float* r = in + row * (long)cols;
  float ss = 0.f;
  for (int c = t; c < cols; c += 256) { float v = r[c]; ss += v * v; }
  red[t] = ss; __syncthreads();
  for (int off = 128; off > 0; off >>= 1) { if (t < off) red[t] += red[t + off]; __syncthreads(); }
  float scale = rsqrtf(red[0] / (float)cols + eps);
  for (int c = t; c < cols; c += 256) out[row * (long)cols + c] = (__bf16)(r[c] * scale * w[c]);
}

// ---------------------------------------------------------------- layernorm -> f32 (1 block/row)
__global__ void layernorm_kernel(const float* __restrict__ in, const float* __restrict__ g,
                                 const float* __restrict__ b, float* __restrict__ out, int cols) {
  __shared__ float red[256];
  long row = blockIdx.x;
  int t = threadIdx.x;
  const float* r = in + row * (long)cols;
  float s = 0.f;
  for (int c = t; c < cols; c += 256) s += r[c];
  red[t] = s; __syncthreads();
  for (int off = 128; off > 0; off >>= 1) { if (t < off) red[t] += red[t + off]; __syncthreads(); }
  float mean = red[0] / (float)cols; __syncthreads();
  float v = 0.f;
  for (int c = t; c < cols; c += 256) { float d = r[c] - mean; v += d * d; }
  red[t] = v; __syncthreads();
  for (int off = 128; off > 0; off >>= 1) { if (t < off) red[t] += red[t + off]; __syncthreads(); }
  float sc = rsqrtf(red[0] / (float)cols + EPS_LN);
  for (int c = t; c < cols; c += 256) out[row * (long)cols + c] = (r[c] - mean) * sc * g[c] + b[c];
}

// ---------------------------------------------------------------- outer[i,j,:] = rows[i,:] + cols[j,:]  (bf16 out)
__global__ void outer_kernel(const float* __restrict__ rows, const float* __restrict__ cols,
                             __bf16* __restrict__ out) {
  long m = blockIdx.x;           // m = i*256 + j
  int i = (int)(m >> 8), j = (int)(m & 255), d = threadIdx.x;
  out[m * DIME + d] = (__bf16)(rows[i * DIME + d] + cols[j * DIME + d]);
}

// ---------------------------------------------------------------- edges += C + rp[i] + cp[j] + bias
__global__ void n2e_epilogue(const float* __restrict__ Cm, const float* __restrict__ rp,
                             const float* __restrict__ cp, const float* __restrict__ bias,
                             float* __restrict__ edges) {
  long m = blockIdx.x;
  int i = (int)(m >> 8), j = (int)(m & 255), d = threadIdx.x;
  long idx = m * DIME + d;
  edges[idx] += Cm[idx] + rp[i * DIME + d] + cp[j * DIME + d] + bias[d];
}

// ---------------------------------------------------------------- bf16 WMMA GEMM
// C[M,N] = A[M,K]@B[K,N]; A,B bf16 global; fp32 accum.
// Block tile 128x64, K-tile 32, 8 waves; each wave owns 32x32 (2x2 WMMA).
// A tile staged via Tensor Data Mover (padded rows: 32 elem + 8 pad = 40),
// B tile staged transposed [col][k] so every fragment is contiguous b128 LDS loads.
// EPI: 0 -> f32 store, 1 -> f32 += residual, 2 -> exact GELU stored as bf16
template <int EPI>
__global__ void gemm_wmma_bf16(const __bf16* __restrict__ A, const __bf16* __restrict__ B,
                               void* __restrict__ Cv, const float* __restrict__ R,
                               int M, int N, int K) {
  __shared__ __align__(16) __bf16 As[128][40];   // 80B row stride (16B aligned)
  __shared__ __align__(16) __bf16 Bst[64][40];   // transposed B tile
  int t = threadIdx.x;
  int lane = t & 31;
  int w = t >> 5;
  int wr = w & 3, wc = w >> 2;
  long row0 = (long)blockIdx.y * 128;
  long col0 = (long)blockIdx.x * 64;

  v8f zero;
  #pragma unroll
  for (int q = 0; q < 8; ++q) zero[q] = 0.f;
  v8f acc[2][2];
  acc[0][0] = zero; acc[0][1] = zero; acc[1][0] = zero; acc[1][1] = zero;

  int lrow   = lane & 15;
  int kbaseA = (lane < 16) ? 0 : 8;    // ISA 16-bit A 16x32 per-lane layout
  int kbaseB = (lane < 16) ? 0 : 8;    // B column-per-lane layout, same K order as A

  for (int k0 = 0; k0 < K; k0 += 32) {
#if HAVE_TDM
    if (t < 32) {
      // D# group0: count=1, lds_addr, 57-bit global addr, type=2
      unsigned lds_a = (unsigned)(uintptr_t)(&As[0][0]);
      unsigned long long ga =
          (unsigned long long)(uintptr_t)A + ((unsigned long long)row0 * (unsigned)K + (unsigned)k0) * 2ull;
      v4u g0;
      g0[0] = 1u;
      g0[1] = lds_a;
      g0[2] = (unsigned)ga;
      g0[3] = (unsigned)(ga >> 32) | 0x80000000u;
      // D# group1: data_size=1(2B), pad_enable, pad_interval=3(16 dw), pad_amount=3(4 dw = 8 bf16)
      v8i g1;
      g1[0] = (int)0x06D10000u;
      g1[1] = (int)(((unsigned)K & 0xFFFFu) << 16);            // tensor_dim0 lo16
      g1[2] = (int)(((unsigned)K >> 16) | (((unsigned)M & 0xFFFFu) << 16));
      g1[3] = (int)(((unsigned)M >> 16) | (32u << 16));        // tile_dim0 = 32
      g1[4] = (int)128;                                        // tile_dim1 = 128, tile_dim2 = 0
      g1[5] = (int)K;                                          // tensor_dim0_stride lo32
      g1[6] = 0;
      g1[7] = 0;
      v4i gz; gz[0] = 0; gz[1] = 0; gz[2] = 0; gz[3] = 0;
#if __clang_major__ >= 23
      v8i gz8; for (int q = 0; q < 8; ++q) gz8[q] = 0;
      __builtin_amdgcn_tensor_load_to_lds(g0, g1, gz, gz, gz8, 0);
#else
      __builtin_amdgcn_tensor_load_to_lds(g0, g1, gz, gz, 0);
#endif
    }
#else
    for (int s = t; s < 1024; s += 256) {         // 128x32 bf16 via b64 loads
      int r = s >> 3, c4 = (s & 7) << 2;
      *(v4bf*)&As[r][c4] = *(const v4bf*)&A[(row0 + r) * (long)K + k0 + c4];
    }
#endif
    // stage B transposed: 32x64 tile -> Bst[col][k]
    for (int s = t; s < 512; s += 256) {
      int r = s >> 4, c4 = (s & 15) << 2;
      v4bf q = *(const v4bf*)&B[(long)(k0 + r) * N + col0 + c4];
      Bst[c4 + 0][r] = q[0];
      Bst[c4 + 1][r] = q[1];
      Bst[c4 + 2][r] = q[2];
      Bst[c4 + 3][r] = q[3];
    }
#if HAVE_TDM
    if (t < 32) __builtin_amdgcn_s_wait_tensorcnt(0);
#endif
    __syncthreads();

    #pragma unroll
    for (int mi = 0; mi < 2; ++mi) {
      union { v16bf v; v8bf h[2]; } fa;
      int arow = wr * 32 + mi * 16 + lrow;
      fa.h[0] = *(const v8bf*)&As[arow][kbaseA];        // K = kbase..kbase+7
      fa.h[1] = *(const v8bf*)&As[arow][kbaseA + 16];   // K = kbase+16..kbase+23
      #pragma unroll
      for (int ni = 0; ni < 2; ++ni) {
        union { v16bf v; v8bf h[2]; } fb;
        int bcol = wc * 32 + ni * 16 + lrow;
        fb.h[0] = *(const v8bf*)&Bst[bcol][kbaseB];
        fb.h[1] = *(const v8bf*)&Bst[bcol][kbaseB + 16];
        acc[mi][ni] = __builtin_amdgcn_wmma_f32_16x16x32_bf16(
            false, fa.v, false, fb.v, (short)0, acc[mi][ni], false, false);
      }
    }
    __syncthreads();
  }

  float*  Cf = (float*)Cv;
  __bf16* Cb = (__bf16*)Cv;
  #pragma unroll
  for (int mi = 0; mi < 2; ++mi)
    #pragma unroll
    for (int ni = 0; ni < 2; ++ni)
      #pragma unroll
      for (int r = 0; r < 8; ++r) {
        long row = row0 + wr * 32 + mi * 16 + r + ((lane >= 16) ? 8 : 0);
        long col = col0 + wc * 32 + ni * 16 + (lane & 15);
        long idx = row * (long)N + col;
        float v = acc[mi][ni][r];
        if (EPI == 1) v += R[idx];
        if (EPI == 2) Cb[idx] = (__bf16)(0.5f * v * (1.0f + erff(v * 0.70710678118654752f)));
        else          Cf[idx] = v;
      }
}

// ---------------------------------------------------------------- relational attention (bf16 out)
__global__ void attention_kernel(const float* __restrict__ nqkv, const float* __restrict__ eqkv,
                                 __bf16* __restrict__ out) {
  __shared__ float p[256];
  __shared__ float red[256];
  int head = blockIdx.x;
  int i    = blockIdx.y;
  int t    = threadIdx.x;
  int j    = t;
  long m   = (long)i * 256 + j;
  const float* nq = nqkv + (long)j * 1536 + head * 64;
  const float* nk = nq + 512;
  const float* eq = eqkv + m * 1536 + head * 64;
  const float* ek = eq + 512;
  float s = 0.f;
  #pragma unroll 8
  for (int d = 0; d < 64; ++d) {
    float q = (nq[d] + eq[d]) * 8.0f;
    float k = nk[d] + ek[d];
    s += q * k;
  }
  red[t] = s; __syncthreads();
  for (int off = 128; off > 0; off >>= 1) { if (t < off) red[t] = fmaxf(red[t], red[t + off]); __syncthreads(); }
  float mx = red[0]; __syncthreads();
  float e = expf(s - mx);
  red[t] = e; __syncthreads();
  for (int off = 128; off > 0; off >>= 1) { if (t < off) red[t] += red[t + off]; __syncthreads(); }
  float sum = red[0]; __syncthreads();
  p[t] = e / sum;
  __syncthreads();
  int d = t & 63;
  int g = t >> 6;
  float acc = 0.f;
  long base_n = (long)(g * 64) * 1536 + 1024 + head * 64 + d;
  long base_e = ((long)i * 256 + g * 64) * 1536 + 1024 + head * 64 + d;
  for (int jj = 0; jj < 64; ++jj)
    acc += p[g * 64 + jj] * (nqkv[base_n + (long)jj * 1536] + eqkv[base_e + (long)jj * 1536]);
  red[t] = acc; __syncthreads();
  if (g == 0)
    out[(long)i * 512 + head * 64 + d] =
        (__bf16)(red[d] + red[64 + d] + red[128 + d] + red[192 + d]);
}

// ---------------------------------------------------------------- launcher
extern "C" void kernel_launch(void* const* d_in, const int* in_sizes, int n_in,
                              void* d_out, int out_size, void* d_ws, size_t ws_size,
                              hipStream_t stream) {
  const int*   x         = (const int*)d_in[0];
  const float* tok       = (const float*)d_in[1];
  const float* pos       = (const float*)d_in[2];
  const float* edge_rows = (const float*)d_in[3];
  const float* edge_cols = (const float*)d_in[4];
  const float* edge_proj = (const float*)d_in[5];
  const float* eln_g     = (const float*)d_in[6];
  const float* eln_b     = (const float*)d_in[7];
  const float* logits_w  = (const float*)d_in[8];

  // workspace carving (256B aligned); peak ~540MB
  char* base = (char*)d_ws;
  auto carve = [&](size_t bytes) { char* p = base; base += (bytes + 255) & ~(size_t)255; return p; };
  float*  h     = (float*)carve((size_t)NND * DIMN * 4);
  float*  nqkv  = (float*)carve((size_t)NND * 1536 * 4);
  float*  rp    = (float*)carve((size_t)NND * DIME * 4);
  float*  cp    = (float*)carve((size_t)NND * DIME * 4);
  float*  edges = (float*)carve((size_t)65536 * DIME * 4);
  float*  big0  = (float*)carve((size_t)65536 * 1536 * 4);  // eqkv f32; aliases below
  __bf16* hnb   = (__bf16*)carve((size_t)NND * DIMN * 2);
  __bf16* bb    = (__bf16*)carve((size_t)65536 * DIME * 2); // normed/outer edges bf16
  __bf16* attnb = (__bf16*)carve((size_t)NND * DIMN * 2);
  __bf16* hffb  = (__bf16*)carve((size_t)NND * 2048 * 2);
  __bf16* hb    = (__bf16*)carve((size_t)NND * DIMN * 2);
  __bf16* wbuf  = (__bf16*)carve((size_t)DIMN * 32000 * 2); // largest weight (logits)
  float*  n2eC  = big0;                                     // alias (sequenced)
  __bf16* ehidb = (__bf16*)big0;                            // alias (sequenced)

  auto cvt = [&](const float* src, __bf16* dst, long n) {
    cvt_bf16_kernel<<<(unsigned)((n + 255) / 256), 256, 0, stream>>>(src, dst, n);
  };
  auto gemm = [&](const __bf16* A, const __bf16* Bm, void* C, const float* Rr,
                  int M, int Nn, int K, int epi) {
    dim3 grid(Nn / 64, M / 128);
    if (epi == 0)      hipLaunchKernelGGL((gemm_wmma_bf16<0>), grid, dim3(256), 0, stream, A, Bm, C, nullptr, M, Nn, K);
    else if (epi == 1) hipLaunchKernelGGL((gemm_wmma_bf16<1>), grid, dim3(256), 0, stream, A, Bm, C, Rr,      M, Nn, K);
    else               hipLaunchKernelGGL((gemm_wmma_bf16<2>), grid, dim3(256), 0, stream, A, Bm, C, nullptr, M, Nn, K);
  };

  // embedding
  embed_kernel<<<(NND * DIMN) / 256, 256, 0, stream>>>(x, tok, pos, h);
  cvt(h, hb, (long)NND * DIMN);

  // edge construction: edges = LN( (rows_i + cols_j) @ edge_proj )
  cvt(edge_rows, wbuf, (long)DIMN * DIME);
  gemm(hb, wbuf, rp, nullptr, NND, DIME, DIMN, 0);
  cvt(edge_cols, wbuf, (long)DIMN * DIME);
  gemm(hb, wbuf, cp, nullptr, NND, DIME, DIMN, 0);
  outer_kernel<<<65536, 256, 0, stream>>>(rp, cp, bb);
  cvt(edge_proj, wbuf, (long)DIME * DIME);
  gemm(bb, wbuf, big0, nullptr, 65536, DIME, DIME, 0);
  layernorm_kernel<<<65536, 256, 0, stream>>>(big0, eln_g, eln_b, edges, DIME);

  for (int L = 0; L < 2; ++L) {
    const float* const* lp = (const float* const*)(d_in + 9 + L * 15);
    const float* attn_norm       = lp[0];
    const float* attn_edges_norm = lp[1];
    const float* w_qkv           = lp[2];
    const float* w_eqkv          = lp[3];
    const float* w_out           = lp[4];
    const float* ff_norm         = lp[5];
    const float* ff_w1           = lp[6];
    const float* ff_w2           = lp[7];
    const float* n2e_node_norm   = lp[8];
    const float* n2e_edge_norm   = lp[9];
    const float* n2e_w           = lp[10];
    const float* n2e_b           = lp[11];
    const float* eff_norm        = lp[12];
    const float* eff_w1          = lp[13];
    const float* eff_w2          = lp[14];

    // --- relational attention ---
    rmsnorm_kernel<<<NND, 256, 0, stream>>>(h, attn_norm, hnb, DIMN, EPS_RMS);
    rmsnorm_kernel<<<65536, 256, 0, stream>>>(edges, attn_edges_norm, bb, DIME, EPS_RMS);
    cvt(w_qkv, wbuf, (long)DIMN * 1536);
    gemm(hnb, wbuf, nqkv, nullptr, NND, 1536, DIMN, 0);
    cvt(w_eqkv, wbuf, (long)DIME * 1536);
    gemm(bb, wbuf, big0, nullptr, 65536, 1536, DIME, 0);   // dominant GEMM (51.5 GFLOP)
    attention_kernel<<<dim3(NHEAD, NND), 256, 0, stream>>>(nqkv, big0, attnb);
    cvt(w_out, wbuf, (long)DIMN * DIMN);
    gemm(attnb, wbuf, h, h, NND, DIMN, DIMN, 1);           // + residual

    // --- node feedforward ---
    rmsnorm_kernel<<<NND, 256, 0, stream>>>(h, ff_norm, hnb, DIMN, EPS_RMS);
    cvt(ff_w1, wbuf, (long)DIMN * 2048);
    gemm(hnb, wbuf, hffb, nullptr, NND, 2048, DIMN, 2);    // GELU -> bf16
    cvt(ff_w2, wbuf, (long)2048 * DIMN);
    gemm(hffb, wbuf, h, h, NND, DIMN, 2048, 1);            // + residual

    // --- nodes-to-edges ---
    rmsnorm_kernel<<<NND, 256, 0, stream>>>(h, n2e_node_norm, hnb, DIMN, EPS_RMS);
    cvt(n2e_w, wbuf, (long)(2 * DIMN + DIME) * DIME);
    gemm(hnb, wbuf,               rp, nullptr, NND, DIME, DIMN, 0);  // rows 0..511
    gemm(hnb, wbuf + 512 * DIME,  cp, nullptr, NND, DIME, DIMN, 0);  // rows 512..1023
    rmsnorm_kernel<<<65536, 256, 0, stream>>>(edges, n2e_edge_norm, bb, DIME, EPS_RMS);
    gemm(bb, wbuf + 1024 * DIME, n2eC, nullptr, 65536, DIME, DIME, 0);
    n2e_epilogue<<<65536, 256, 0, stream>>>(n2eC, rp, cp, n2e_b, edges);

    // --- edge feedforward ---
    rmsnorm_kernel<<<65536, 256, 0, stream>>>(edges, eff_norm, bb, DIME, EPS_RMS);
    cvt(eff_w1, wbuf, (long)DIME * 1024);
    gemm(bb, wbuf, ehidb, nullptr, 65536, 1024, DIME, 2);  // GELU -> bf16 (aliases big0)
    cvt(eff_w2, wbuf, (long)1024 * DIME);
    gemm(ehidb, wbuf, edges, edges, 65536, DIME, 1024, 1); // + residual
  }

  // logits
  cvt(h, hb, (long)NND * DIMN);
  cvt(logits_w, wbuf, (long)DIMN * 32000);
  gemm(hb, wbuf, (float*)d_out, nullptr, NND, 32000, DIMN, 0);
}
